// Conv2d_int_83425444758205
// MI455X (gfx1250) — 8-bit matrix operand order repaired and checked by reading, compiled, not run on hardware
//
#include <hip/hip_runtime.h>
#include <stdint.h>

typedef __attribute__((ext_vector_type(8))) int v8i;
typedef int v4i_t __attribute__((vector_size(16)));

#define GAS __attribute__((address_space(1)))
#define LAS __attribute__((address_space(3)))

static constexpr int BATCH = 16;
static constexpr int CIN   = 256;
static constexpr int COUT  = 256;
static constexpr int IH    = 56;
static constexpr int IW    = 56;
static constexpr int HWPIX = IH * IW;
static constexpr int NPIX  = BATCH * HWPIX;
static constexpr int TAPS  = 9;
static constexpr int KDIM  = TAPS * CIN;

static constexpr int PIX_TILE = 128;
static constexpr int CO_TILE  = 64;
static constexpr int KC       = 64;
static constexpr int NSTEP    = KDIM / KC;

#if __has_builtin(__builtin_amdgcn_global_load_async_to_lds_b128)
#define HAVE_ASYNC_LDS 1
#else
#define HAVE_ASYNC_LDS 0
#endif

__device__ __forceinline__ void copy16(const unsigned char* g, unsigned char* l) {
#if HAVE_ASYNC_LDS
    __builtin_amdgcn_global_load_async_to_lds_b128(
        (GAS v4i_t*)(uintptr_t)g,
        (LAS v4i_t*)(unsigned int)(uintptr_t)l,
         0,  0);
#else
    *(uint4*)l = *(const uint4*)g;
#endif
}

__device__ __forceinline__ void wait_async3() {
#if HAVE_ASYNC_LDS
#if __has_builtin(__builtin_amdgcn_s_wait_asynccnt)
    __builtin_amdgcn_s_wait_asynccnt(3);
#else
    asm volatile("s_wait_asynccnt 0x3" ::: "memory");
#endif
#endif
}
__device__ __forceinline__ void wait_async0() {
#if HAVE_ASYNC_LDS
#if __has_builtin(__builtin_amdgcn_s_wait_asynccnt)
    __builtin_amdgcn_s_wait_asynccnt(0);
#else
    asm volatile("s_wait_asynccnt 0x0" ::: "memory");
#endif
#endif
}

__global__ void pack_x_nhwc_kernel(const int* __restrict__ src,
                                   unsigned int* __restrict__ dst) {
    int idx = blockIdx.x * blockDim.x + threadIdx.x;
    int sp  = idx >> 6;
    int c4  = idx & 63;
    int b   = sp / HWPIX;
    int rem = sp - b * HWPIX;
    const int* s = src + (size_t)(b * CIN + c4 * 4) * HWPIX + rem;
    unsigned int p = (unsigned int)(s[0] & 0xff)
                   | ((unsigned int)(s[HWPIX]     & 0xff) << 8)
                   | ((unsigned int)(s[2 * HWPIX] & 0xff) << 16)
                   | ((unsigned int)(s[3 * HWPIX] & 0xff) << 24);
    dst[idx] = p;
}

__global__ void pack_w_kernel(const int* __restrict__ src,
                              unsigned int* __restrict__ dst,
                              unsigned int* __restrict__ zpad) {
    int idx = blockIdx.x * blockDim.x + threadIdx.x;
    if (blockIdx.x == 0 && threadIdx.x < 64) zpad[threadIdx.x] = 0u;
    int co  = idx / (KDIM / 4);
    int r   = idx - co * (KDIM / 4);
    int tap = r >> 6;
    int ci4 = r & 63;
    const int* s = src + ((size_t)co * CIN + ci4 * 4) * TAPS + tap;
    unsigned int p = (unsigned int)(s[0]  & 0xff)
                   | ((unsigned int)(s[TAPS]     & 0xff) << 8)
                   | ((unsigned int)(s[2 * TAPS] & 0xff) << 16)
                   | ((unsigned int)(s[3 * TAPS] & 0xff) << 24);
    dst[idx] = p;
}

__global__ __launch_bounds__(256)
void conv_wmma_iu8_kernel(const unsigned char* __restrict__ xq,
                          const unsigned char* __restrict__ wq,
                          const unsigned char* __restrict__ zpad,
                          const int* __restrict__ bias,
                          int* __restrict__ out) {
    __shared__ unsigned char lds_w[2][CO_TILE * KC];
    __shared__ unsigned char lds_x[2][PIX_TILE * KC];

    const int t    = threadIdx.x;
    const int lane = t & 31;
    const int wave = t >> 5;
    const int wrow = wave >> 2;
    const int wcol = wave & 3;

    const int p0  = blockIdx.x * PIX_TILE;
    const int co0 = blockIdx.y * CO_TILE;

    const int wl_row = t >> 2;
    const int wl_off = (t & 3) * 16;
    const int xl_row = t >> 1;
    const int xl_off = (t & 1) * 32;

    const int p   = p0 + xl_row;
    const int bb  = p / HWPIX;
    const int rem = p - bb * HWPIX;
    const int oh  = rem / IW;
    const int ow  = rem - oh * IW;
    const unsigned char* xpix = xq + (size_t)(bb * IH) * IW * CIN;
    const unsigned char* zsrc = zpad + xl_off;

    const unsigned char* w_ptr = wq + (size_t)(co0 + wl_row) * KDIM + wl_off;
    int kh = 0, kw = 0, jci = 0;
    const unsigned char* x_ptr;
    {
        const int ihh = oh - 1, iww = ow - 1;
        const bool valid =
            ((unsigned)ihh < (unsigned)IH) && ((unsigned)iww < (unsigned)IW);
        x_ptr = valid ? xpix + ((size_t)ihh * IW + iww) * CIN + xl_off : zsrc;
    }

    auto stage = [&](int buf) {
        copy16(x_ptr,      &lds_x[buf][xl_row * KC + xl_off]);
        copy16(x_ptr + 16, &lds_x[buf][xl_row * KC + xl_off + 16]);
        copy16(w_ptr,      &lds_w[buf][wl_row * KC + wl_off]);
    };
    auto advance = [&]() {
        w_ptr += KC;
        if (++jci == CIN / KC) {
            jci = 0;
            if (++kw == 3) { kw = 0; ++kh; }
            const int ihh = oh + kh - 1, iww = ow + kw - 1;
            const bool valid =
                ((unsigned)ihh < (unsigned)IH) && ((unsigned)iww < (unsigned)IW);
            x_ptr = valid ? xpix + ((size_t)ihh * IW + iww) * CIN + xl_off : zsrc;
        } else {
            x_ptr += KC;
        }
    };

    v8i acc00 = {}, acc01 = {}, acc10 = {}, acc11 = {};

    const int M  = lane & 15;
    const int hi = lane >> 4;

    auto compute = [&](int buf) {
        const unsigned int* wf0 = (const unsigned int*)&lds_w[buf][(wrow * 32 +      M) * KC];
        const unsigned int* wf1 = (const unsigned int*)&lds_w[buf][(wrow * 32 + 16 + M) * KC];
        const unsigned int* xf0 = (const unsigned int*)&lds_x[buf][(wcol * 32 +      M) * KC];
        const unsigned int* xf1 = (const unsigned int*)&lds_x[buf][(wcol * 32 + 16 + M) * KC];
        v8i a0 = {}, a1 = {}, b0 = {}, b1 = {};
#pragma unroll
        for (int v = 0; v < 8; ++v) {
            const int ka = ((v >> 1) * 16 + (v & 1) * 4 + hi * 8)  >> 2;
            const int kb = ((v >> 1) * 16 + (v & 1) * 4 + hi * 8)  >> 2;
            a0[v] = (int)wf0[ka];
            a1[v] = (int)wf1[ka];
            b0[v] = (int)xf0[kb];
            b1[v] = (int)xf1[kb];
        }
        acc00 = __builtin_amdgcn_wmma_i32_16x16x64_iu8(false, a0, false, b0, acc00, false, false);
        acc01 = __builtin_amdgcn_wmma_i32_16x16x64_iu8(false, a0, false, b1, acc01, false, false);
        acc10 = __builtin_amdgcn_wmma_i32_16x16x64_iu8(false, a1, false, b0, acc10, false, false);
        acc11 = __builtin_amdgcn_wmma_i32_16x16x64_iu8(false, a1, false, b1, acc11, false, false);
    };

    stage(0);
    advance();
    for (int s = 0; s < NSTEP - 1; ++s) {
        stage((s + 1) & 1);
        advance();
        wait_async3();
        __syncthreads();
        compute(s & 1);
        __syncthreads();
    }
    wait_async0();
    __syncthreads();
    compute((NSTEP - 1) & 1);

    const int N    = lane & 15;
    const int pixA = p0 + wcol * 32 + N;
    const int pixB = pixA + 16;
    const int obA  = pixA / HWPIX, orA = pixA - obA * HWPIX;
    const int obB  = pixB / HWPIX, orB = pixB - obB * HWPIX;
#pragma unroll
    for (int e = 0; e < 8; ++e) {
        const int co_lo = co0 + wrow * 32 + e + hi * 8;
        const int co_hi = co_lo + 16;
        out[(size_t)(obA * COUT + co_lo) * HWPIX + orA] = acc00[e] + bias[co_lo];
        out[(size_t)(obB * COUT + co_lo) * HWPIX + orB] = acc01[e] + bias[co_lo];
        out[(size_t)(obA * COUT + co_hi) * HWPIX + orA] = acc10[e] + bias[co_hi];
        out[(size_t)(obB * COUT + co_hi) * HWPIX + orB] = acc11[e] + bias[co_hi];
    }
}

extern "C" void kernel_launch(void* const* d_in, const int* in_sizes, int n_in,
                              void* d_out, int out_size, void* d_ws, size_t ws_size,
                              hipStream_t stream) {
    const int* x    = (const int*)d_in[0];
    const int* w    = (const int*)d_in[1];
    const int* bias = (const int*)d_in[2];
    int* out        = (int*)d_out;

    unsigned char* xq   = (unsigned char*)d_ws;
    unsigned char* wq   = xq + (size_t)NPIX * CIN;
    unsigned char* zpad = wq + (size_t)COUT * KDIM;

    const int nx4 = NPIX * (CIN / 4);
    const int nw4 = COUT * (KDIM / 4);
    pack_x_nhwc_kernel<<<nx4 / 256, 256, 0, stream>>>(x, (unsigned int*)xq);
    pack_w_kernel<<<nw4 / 256, 256, 0, stream>>>(w, (unsigned int*)wq,
                                                 (unsigned int*)zpad);

    dim3 grid(NPIX / PIX_TILE, COUT / CO_TILE);
    conv_wmma_iu8_kernel<<<grid, 256, 0, stream>>>(xq, wq, zpad, bias, out);
}
